// BilinearInteraction1_8821862826066
// MI455X (gfx1250) — matrix operand order repaired and checked by reading, not compiled, not run on hardware
//
#include <hip/hip_runtime.h>

#define NF      32
#define BATCHN  4096
#define EMB     64
#define PAIRS   496
#define TILE    128
#define THREADS 256

typedef __attribute__((ext_vector_type(16))) __bf16 v16bf;
typedef __attribute__((ext_vector_type(8)))  __bf16 v8bf;
typedef __attribute__((ext_vector_type(4)))  __bf16 v4bf;
typedef __attribute__((ext_vector_type(8)))  float  v8f;

// float -> bf16 round-to-nearest-even via integer ops
__device__ __forceinline__ __bf16 tobf(float x) {
  unsigned u = __builtin_bit_cast(unsigned, x);
  u += 0x7FFFu + ((u >> 16) & 1u);
  unsigned short hs = (unsigned short)(u >> 16);
  return __builtin_bit_cast(__bf16, hs);
}
__device__ __forceinline__ float bf2f(__bf16 b) {
  unsigned short hs = __builtin_bit_cast(unsigned short, b);
  unsigned u = ((unsigned)hs) << 16;
  return __builtin_bit_cast(float, u);
}

union UF { v16bf v; v8bf h[2]; };

// ---- CDNA5 async memory->LDS copy (16B per lane), tracked by ASYNCcnt ----
// Flat shared pointers on amdgcn are {aperture_hi32, lds_offset32}; truncation
// to 32 bits yields the LDS byte address the instruction wants.
__device__ __forceinline__ void async_b128(const void* g, void* l) {
  asm volatile("global_load_async_to_lds_b128 %0, %1, off"
               :: "v"((unsigned)(uintptr_t)l), "v"(g)
               : "memory");
}
__device__ __forceinline__ void wait_async0() {
  asm volatile("s_wait_asynccnt 0x0" ::: "memory");
}

// ---------------- prep kernel 1: split X -> Xh, Xl (same row-major layout) ----------------
__global__ __launch_bounds__(THREADS)
void split_x_kernel(const float* __restrict__ X, __bf16* __restrict__ Xh,
                    __bf16* __restrict__ Xl, int n4) {
  int q = blockIdx.x * blockDim.x + threadIdx.x;  // float4 index
  if (q >= n4) return;
  float4 v = ((const float4*)X)[q];
  float t[4] = { v.x, v.y, v.z, v.w };
  v4bf hs, ls;
  #pragma unroll
  for (int c = 0; c < 4; ++c) {
    __bf16 h = tobf(t[c]);
    hs[c] = h;
    ls[c] = tobf(t[c] - bf2f(h));
  }
  *(v4bf*)(Xh + (size_t)q * 4) = hs;
  *(v4bf*)(Xl + (size_t)q * 4) = ls;
}

// ------------- prep kernel 2: transpose + split W -> WTh, WTl laid out [p][l][k] -------------
__global__ __launch_bounds__(THREADS)
void split_wt_kernel(const float* __restrict__ W, __bf16* __restrict__ WTh,
                     __bf16* __restrict__ WTl) {
  __shared__ float sT[EMB][EMB + 1];
  const int p   = blockIdx.x;
  const int tid = threadIdx.x;
  const float* Wp = W + (size_t)p * (EMB * EMB);
  #pragma unroll
  for (int s = 0; s < 16; ++s) {
    int e = tid + THREADS * s;         // coalesced read [k][l]
    int k = e >> 6, l = e & 63;
    sT[l][k] = Wp[e];
  }
  __syncthreads();
  __bf16* oh = WTh + (size_t)p * (EMB * EMB);
  __bf16* ol = WTl + (size_t)p * (EMB * EMB);
  #pragma unroll
  for (int s = 0; s < 16; ++s) {
    int e = tid + THREADS * s;         // coalesced write [l][k]
    int l = e >> 6, k = e & 63;
    float w  = sT[l][k];
    __bf16 h = tobf(w);
    oh[e] = h;
    ol[e] = tobf(w - bf2f(h));
  }
}

// ---------------- main kernel: bf16 pre-split, async LDS staging, A direct from global ----------------
__global__ __launch_bounds__(THREADS)
void bilinear_main(const __bf16* __restrict__ Xh, const __bf16* __restrict__ Xl,
                   const __bf16* __restrict__ WTh, const __bf16* __restrict__ WTl,
                   const float* __restrict__ X,   // fp32 originals, for Vj epilogue
                   float* __restrict__ out) {
  __shared__ __bf16 sWh[EMB][72];      // W^T hi: [l][k], padded 144B rows
  __shared__ __bf16 sWl[EMB][72];
  __shared__ float  sVj[TILE][68];     // Vj fp32, padded 272B rows

  const int p  = blockIdx.y;
  const int b0 = blockIdx.x * TILE;

  int fi = 0, fj = 1, rem = p;
  for (int i = 0; i < NF - 1; ++i) {
    int cnt = NF - 1 - i;
    if (rem < cnt) { fi = i; fj = i + 1 + rem; break; }
    rem -= cnt;
  }

  const int tid  = threadIdx.x;
  const int wave = tid >> 5;
  const int lane = tid & 31;
  const int lr   = lane & 15;
  const int half = lane >> 4;

  // ---- kick off async LDS staging first (no VGPR round trip) ----
  {
    const int l  = tid >> 2;
    const int k0 = (tid & 3) * 16;
    const size_t g = (size_t)p * (EMB * EMB) + (size_t)tid * 16;
    async_b128(WTh + g,     &sWh[l][k0]);
    async_b128(WTh + g + 8, &sWh[l][k0 + 8]);
    async_b128(WTl + g,     &sWl[l][k0]);
    async_b128(WTl + g + 8, &sWl[l][k0 + 8]);
  }
  {
    const float* Xj = X + (size_t)fj * (BATCHN * EMB);
    #pragma unroll
    for (int s = 0; s < 8; ++s) {
      int q = tid + THREADS * s;
      int f = q << 2;
      int r = f >> 6, k = f & 63;
      async_b128(Xj + (size_t)(b0 + r) * EMB + k, &sVj[r][k]);
    }
  }

  // ---- A fragments straight from global (overlap the async fill) ----
  // A 16x32 bf16 layout: lane<16 holds K {0..7,16..23}; lane>=16 holds K {8..15,24..31}
  const size_t arow = (size_t)fi * (BATCHN * EMB) + (size_t)(b0 + 16 * wave + lr) * EMB;
  const __bf16* xh = Xh + arow;
  const __bf16* xl = Xl + arow;
  UF ah[2], al[2];
  #pragma unroll
  for (int ks = 0; ks < 2; ++ks) {
    const int ab = 32 * ks + 8 * half;
    ah[ks].h[0] = *(const v8bf*)(xh + ab);
    ah[ks].h[1] = *(const v8bf*)(xh + ab + 16);
    al[ks].h[0] = *(const v8bf*)(xl + ab);
    al[ks].h[1] = *(const v8bf*)(xl + ab + 16);
  }

  wait_async0();       // our async writes into LDS are done
  __syncthreads();     // everyone's are

  v8f acc[4];
  #pragma unroll
  for (int n = 0; n < 4; ++n)
    #pragma unroll
    for (int r = 0; r < 8; ++r) acc[n][r] = 0.0f;

  #pragma unroll
  for (int ks = 0; ks < 2; ++ks) {
    #pragma unroll
    for (int n = 0; n < 4; ++n) {
      // B 32x16 bf16: lane n holds column N, K {0..7,16..23} (lanes<16) / {8..15,24..31} (lanes>=16)
      const int bcol = 16 * n + lr;
      const int kb   = 32 * ks + 8 * half;
      UF bh, bl;
      bh.h[0] = *(const v8bf*)&sWh[bcol][kb];
      bh.h[1] = *(const v8bf*)&sWh[bcol][kb + 16];
      bl.h[0] = *(const v8bf*)&sWl[bcol][kb];
      bl.h[1] = *(const v8bf*)&sWl[bcol][kb + 16];

      acc[n] = __builtin_amdgcn_wmma_f32_16x16x32_bf16(false, ah[ks].v, false, bh.v,
                                                       (short)0, acc[n], false, false);
      acc[n] = __builtin_amdgcn_wmma_f32_16x16x32_bf16(false, ah[ks].v, false, bl.v,
                                                       (short)0, acc[n], false, false);
      acc[n] = __builtin_amdgcn_wmma_f32_16x16x32_bf16(false, al[ks].v, false, bh.v,
                                                       (short)0, acc[n], false, false);
    }
  }

  // ---- epilogue: out[b,p] = sum_l C[b,l] * Vj[b,l] ----
  float s[8];
  #pragma unroll
  for (int r = 0; r < 8; ++r) s[r] = 0.0f;
  #pragma unroll
  for (int n = 0; n < 4; ++n) {
    const int col = 16 * n + lr;
    #pragma unroll
    for (int r = 0; r < 8; ++r)
      s[r] += acc[n][r] * sVj[16 * wave + 8 * half + r][col];
  }
  #pragma unroll
  for (int m = 1; m < 16; m <<= 1)
    #pragma unroll
    for (int r = 0; r < 8; ++r)
      s[r] += __shfl_xor(s[r], m, 32);

  if (lr == 0) {
    #pragma unroll
    for (int r = 0; r < 8; ++r) {
      const int gb = b0 + 16 * wave + 8 * half + r;
      out[(size_t)gb * PAIRS + p] = s[r];
    }
  }
}

// ---------------- fallback (self-contained, no workspace) ----------------
__global__ __launch_bounds__(THREADS)
void bilinear_fallback(const float* __restrict__ X, const float* __restrict__ W,
                       float* __restrict__ out) {
  __shared__ __bf16 sAh[TILE][72];
  __shared__ __bf16 sAl[TILE][72];
  __shared__ __bf16 sWh[EMB][72];
  __shared__ __bf16 sWl[EMB][72];
  __shared__ float  sVj[TILE][68];

  const int p  = blockIdx.y;
  const int b0 = blockIdx.x * TILE;
  int fi = 0, fj = 1, rem = p;
  for (int i = 0; i < NF - 1; ++i) {
    int cnt = NF - 1 - i;
    if (rem < cnt) { fi = i; fj = i + 1 + rem; break; }
    rem -= cnt;
  }
  const int tid = threadIdx.x;
  const float* Xi = X + (size_t)fi * (BATCHN * EMB);
  const float* Xj = X + (size_t)fj * (BATCHN * EMB);
  const float* Wp = W + (size_t)p  * (EMB * EMB);

  #pragma unroll
  for (int s = 0; s < 16; ++s) {
    int e = tid + THREADS * s;
    int k = e >> 6, l = e & 63;
    float w  = Wp[e];
    __bf16 h = tobf(w);
    sWh[l][k] = h;
    sWl[l][k] = tobf(w - bf2f(h));
  }
  #pragma unroll
  for (int s = 0; s < 8; ++s) {
    int q = tid + THREADS * s;
    int f = q << 2;
    int r = f >> 6, k = f & 63;
    const size_t goff = (size_t)(b0 + r) * EMB + k;
    float4 vi = *(const float4*)(Xi + goff);
    float4 vj = *(const float4*)(Xj + goff);
    float tv[4] = { vi.x, vi.y, vi.z, vi.w };
    #pragma unroll
    for (int c = 0; c < 4; ++c) {
      __bf16 h = tobf(tv[c]);
      sAh[r][k + c] = h;
      sAl[r][k + c] = tobf(tv[c] - bf2f(h));
    }
    *(float4*)&sVj[r][k] = vj;
  }
  __syncthreads();

  const int wave = tid >> 5, lane = tid & 31, lr = lane & 15, half = lane >> 4;
  v8f acc[4];
  #pragma unroll
  for (int n = 0; n < 4; ++n)
    #pragma unroll
    for (int r = 0; r < 8; ++r) acc[n][r] = 0.0f;

  #pragma unroll
  for (int ks = 0; ks < 2; ++ks) {
    const int arow = 16 * wave + lr;
    const int abase = 32 * ks + 8 * half;
    UF ah, al;
    ah.h[0] = *(const v8bf*)&sAh[arow][abase];
    ah.h[1] = *(const v8bf*)&sAh[arow][abase + 16];
    al.h[0] = *(const v8bf*)&sAl[arow][abase];
    al.h[1] = *(const v8bf*)&sAl[arow][abase + 16];
    #pragma unroll
    for (int n = 0; n < 4; ++n) {
      const int bcol = 16 * n + lr;
      const int kb   = 32 * ks + 8 * half;
      UF bh, bl;
      bh.h[0] = *(const v8bf*)&sWh[bcol][kb];
      bh.h[1] = *(const v8bf*)&sWh[bcol][kb + 16];
      bl.h[0] = *(const v8bf*)&sWl[bcol][kb];
      bl.h[1] = *(const v8bf*)&sWl[bcol][kb + 16];
      acc[n] = __builtin_amdgcn_wmma_f32_16x16x32_bf16(false, ah.v, false, bh.v,
                                                       (short)0, acc[n], false, false);
      acc[n] = __builtin_amdgcn_wmma_f32_16x16x32_bf16(false, ah.v, false, bl.v,
                                                       (short)0, acc[n], false, false);
      acc[n] = __builtin_amdgcn_wmma_f32_16x16x32_bf16(false, al.v, false, bh.v,
                                                       (short)0, acc[n], false, false);
    }
  }
  float s[8];
  #pragma unroll
  for (int r = 0; r < 8; ++r) s[r] = 0.0f;
  #pragma unroll
  for (int n = 0; n < 4; ++n) {
    const int col = 16 * n + lr;
    #pragma unroll
    for (int r = 0; r < 8; ++r)
      s[r] += acc[n][r] * sVj[16 * wave + 8 * half + r][col];
  }
  #pragma unroll
  for (int m = 1; m < 16; m <<= 1)
    #pragma unroll
    for (int r = 0; r < 8; ++r)
      s[r] += __shfl_xor(s[r], m, 32);
  if (lr == 0) {
    #pragma unroll
    for (int r = 0; r < 8; ++r) {
      const int gb = b0 + 16 * wave + 8 * half + r;
      out[(size_t)gb * PAIRS + p] = s[r];
    }
  }
}

extern "C" void kernel_launch(void* const* d_in, const int* in_sizes, int n_in,
                              void* d_out, int out_size, void* d_ws, size_t ws_size,
                              hipStream_t stream) {
  (void)in_sizes; (void)n_in; (void)out_size;
  const float* X = (const float*)d_in[0];
  const float* W = (const float*)d_in[1];
  float* out = (float*)d_out;

  const size_t nX = (size_t)NF * BATCHN * EMB;             // 8,388,608
  const size_t nW = (size_t)PAIRS * EMB * EMB;             // 2,031,616
  const size_t need = (2 * nX + 2 * nW) * sizeof(__bf16);  // ~41.7 MB

  dim3 grid(BATCHN / TILE, PAIRS);

  if (ws_size >= need && d_ws != nullptr) {
    __bf16* Xh  = (__bf16*)d_ws;
    __bf16* Xl  = Xh + nX;
    __bf16* WTh = Xl + nX;
    __bf16* WTl = WTh + nW;

    int n4 = (int)(nX / 4);
    split_x_kernel<<<(n4 + THREADS - 1) / THREADS, THREADS, 0, stream>>>(X, Xh, Xl, n4);
    split_wt_kernel<<<PAIRS, THREADS, 0, stream>>>(W, WTh, WTl);
    bilinear_main<<<grid, THREADS, 0, stream>>>(Xh, Xl, WTh, WTl, X, out);
  } else {
    bilinear_fallback<<<grid, THREADS, 0, stream>>>(X, W, out);
  }
}
